// LSTransformerEncoderLayer_67894843015649
// MI455X (gfx1250) — matrix operand order repaired and checked by reading, not compiled, not run on hardware
//
#include <hip/hip_runtime.h>

#define HS 1024
#define IMS 4096
#define NHEAD 16
#define HEAD_DIM 64

typedef _Float16 v16h __attribute__((ext_vector_type(16)));
typedef float    v8f  __attribute__((ext_vector_type(8)));
typedef uint32_t u32x4 __attribute__((ext_vector_type(4)));
typedef int      i32x4 __attribute__((ext_vector_type(4)));
typedef int      i32x8 __attribute__((ext_vector_type(8)));

#define WMMA_F16(a, b, c) \
  __builtin_amdgcn_wmma_f32_16x16x32_f16(false, (a), false, (b), (short)0, (c), false, false)

#if defined(__AMDGCN__) && __has_builtin(__builtin_amdgcn_tensor_load_to_lds) && \
    __has_builtin(__builtin_amdgcn_s_wait_tensorcnt)
#define USE_TDM 1
#pragma message("gfx1250 device pass: TDM tensor_load_to_lds path ENABLED")
#if __has_include(<hip/amd_detail/amd_gfx1250_TDM.h>)
#define TDM_ARGS6 1
#else
#define TDM_ARGS6 0
#endif
#else
#define USE_TDM 0
#pragma message("pass without TDM builtins: synchronous staging fallback")
#endif

// ---------------------------------------------------------------------------
// Fragment loaders per CDNA5 ISA 7.12.2 VGPR layouts (wave32, 16x16x32 f16).
// A (16x32, MxK): lanes 0-15 -> M=lane, K-halves {0..7,16..23}; lanes 16-31 ->
//                 same M, K-halves {8..15,24..31}; 2 halves per dword.
// B (32x16, KxN): lane%16 = N; K-halves same as A ({0..7,16..23} / {8..15,24..31}).
// Tiles stored [row][k] so both loaders are packed dword reads.
// ---------------------------------------------------------------------------
__device__ __forceinline__ v16h frag_ld_a(const _Float16* base, int lane, int strideH) {
  const int m  = lane & 15;
  const int kd = (lane & 16) ? 4 : 0;
  const uint32_t* p = (const uint32_t*)(base + (size_t)m * strideH);
  union { v16h h; uint32_t u[8]; } r;
#pragma unroll
  for (int j = 0; j < 4; ++j) { r.u[j] = p[kd + j]; r.u[4 + j] = p[kd + 8 + j]; }
  return r.h;
}

__device__ __forceinline__ v16h frag_ld_b(const _Float16* base, int lane, int strideH) {
  const int n  = lane & 15;
  const int kd = (lane & 16) ? 4 : 0;
  const uint32_t* p = (const uint32_t*)(base + (size_t)n * strideH);
  union { v16h h; uint32_t u[8]; } r;
#pragma unroll
  for (int j = 0; j < 4; ++j) { r.u[j] = p[kd + j]; r.u[4 + j] = p[kd + 8 + j]; }
  return r.h;
}

#if USE_TDM
// ---------------------------------------------------------------------------
// Tensor Data Mover: load a 2D tile (tileCols f16 x tileRows) from a row-major
// tensor (row stride = strideElems) into LDS, inserting (padAmtCode+1) dwords
// of LDS padding after every (1<<padIntvCode) qwords (ISA ch8 D# layout).
// ---------------------------------------------------------------------------
__device__ __forceinline__ void tdm_load_2d(const _Float16* g, uint32_t lds_off,
                                            uint32_t strideElems, uint32_t nrowsTensor,
                                            uint32_t tileCols, uint32_t tileRows,
                                            uint32_t padIntvCode, uint32_t padAmtCode) {
  const uint64_t ga = (uint64_t)(uintptr_t)g;
  u32x4 g0;
  g0[0] = 1u;                                   // count=1, user mode, no gather
  g0[1] = lds_off;                              // lds_addr (bytes)
  g0[2] = (uint32_t)ga;                         // global_addr[31:0]
  g0[3] = ((uint32_t)(ga >> 32) & 0x01FFFFFFu) | (2u << 30);  // addr[56:32], type=2
  i32x8 g1;
  g1[0] = (int)((1u << 16) | (1u << 20) | (padIntvCode << 22) | (padAmtCode << 25));
  g1[1] = (int)((strideElems & 0xFFFFu) << 16);                         // tensor_dim0 lo
  g1[2] = (int)(((strideElems >> 16) & 0xFFFFu) | ((nrowsTensor & 0xFFFFu) << 16));
  g1[3] = (int)(((nrowsTensor >> 16) & 0xFFFFu) | ((tileCols & 0xFFFFu) << 16));
  g1[4] = (int)(tileRows & 0xFFFFu);            // tile_dim1; tile_dim2 = 0
  g1[5] = (int)strideElems;                     // tensor_dim0_stride lo
  g1[6] = 0;                                    // stride hi, dim1_stride lo
  g1[7] = 0;
  i32x4 z4;
  z4[0] = 0; z4[1] = 0; z4[2] = 0; z4[3] = 0;
#if TDM_ARGS6
  i32x8 z8;
#pragma unroll
  for (int i = 0; i < 8; ++i) z8[i] = 0;
  __builtin_amdgcn_tensor_load_to_lds(g0, g1, z4, z4, z8, 0);
#else
  __builtin_amdgcn_tensor_load_to_lds(g0, g1, z4, z4, 0);
#endif
}
#endif  // USE_TDM

// ---------------------------------------------------------------------------
// fp32 -> f16 elementwise convert
// ---------------------------------------------------------------------------
__global__ __launch_bounds__(256) void cvt_f32_to_f16(const float* __restrict__ in,
                                                      _Float16* __restrict__ out, size_t n) {
  size_t i = (size_t)blockIdx.x * blockDim.x + threadIdx.x;
  size_t stride = (size_t)gridDim.x * blockDim.x;
  for (; i < n; i += stride) out[i] = (_Float16)in[i];
}

// ---------------------------------------------------------------------------
// LayerNorm over H=1024, output f16.  One 256-thread block per row.
// ---------------------------------------------------------------------------
__global__ __launch_bounds__(256) void ln_to_f16(const float* __restrict__ x,
                                                 const float* __restrict__ g,
                                                 const float* __restrict__ b,
                                                 _Float16* __restrict__ out) {
  const int row = blockIdx.x;
  const int tid = threadIdx.x;
  const float* xr = x + (size_t)row * HS;
  float4 v = *(const float4*)(xr + tid * 4);
  float s = v.x + v.y + v.z + v.w;
  float s2 = v.x * v.x + v.y * v.y + v.z * v.z + v.w * v.w;
#pragma unroll
  for (int off = 1; off < 32; off <<= 1) {
    s += __shfl_xor(s, off, 32);
    s2 += __shfl_xor(s2, off, 32);
  }
  __shared__ float red[16];
  const int wave = tid >> 5, lane = tid & 31;
  if (lane == 0) { red[wave] = s; red[8 + wave] = s2; }
  __syncthreads();
  float ts = 0.f, ts2 = 0.f;
#pragma unroll
  for (int w = 0; w < 8; ++w) { ts += red[w]; ts2 += red[8 + w]; }
  const float mu = ts * (1.0f / HS);
  const float var = ts2 * (1.0f / HS) - mu * mu;
  const float rs = rsqrtf(var + 1e-5f);
  _Float16* orow = out + (size_t)row * HS;
  const int i0 = tid * 4;
  float4 gg = *(const float4*)(g + i0);
  float4 bb = *(const float4*)(b + i0);
  orow[i0 + 0] = (_Float16)((v.x - mu) * rs * gg.x + bb.x);
  orow[i0 + 1] = (_Float16)((v.y - mu) * rs * gg.y + bb.y);
  orow[i0 + 2] = (_Float16)((v.z - mu) * rs * gg.z + bb.z);
  orow[i0 + 3] = (_Float16)((v.w - mu) * rs * gg.w + bb.w);
}

// ---------------------------------------------------------------------------
// Tiled WMMA GEMM:  out[m][n] = sum_k A[m][k] * W[n][k]  (+bias +relu +res)
// Block tile 128x256x32, 8 waves in 2x4, each wave -> 64x64 (4x4 WMMA tiles).
// Double-buffered LDS fed by the Tensor Data Mover (one tensor_load_to_lds per
// tile per K-step, TENSORcnt-pipelined); synchronous fallback otherwise.
// ---------------------------------------------------------------------------
#define BM 128
#define BN 256
#define BK 32
#define AST 40  // padded LDS row stride in halves: 16 dwords data + 4 dwords pad

template <bool RELU, bool RES, bool OUT16>
__global__ __launch_bounds__(256) void gemm_f16(const _Float16* __restrict__ A,
                                                const _Float16* __restrict__ W,
                                                const float* __restrict__ bias,
                                                const float* __restrict__ res,
                                                _Float16* __restrict__ out16,
                                                float* __restrict__ out32,
                                                int M, int N, int K) {
  __shared__ __align__(16) _Float16 As[2 * BM * AST];
  __shared__ __align__(16) _Float16 Bs[2 * BN * AST];
  const int tid = threadIdx.x;
  const int lane = tid & 31;
  const int wave = tid >> 5;
  const int wm = wave & 1;   // 2 wave-rows of 64
  const int wn = wave >> 1;  // 4 wave-cols of 64
  const size_t rowBase = (size_t)blockIdx.y * BM;
  const size_t colBase = (size_t)blockIdx.x * BN;
  const int niter = K / BK;

  v8f acc[4][4];
#pragma unroll
  for (int mt = 0; mt < 4; ++mt)
#pragma unroll
    for (int nt = 0; nt < 4; ++nt)
#pragma unroll
      for (int v = 0; v < 8; ++v) acc[mt][nt][v] = 0.0f;

#if USE_TDM
  // pad codes: interval 3 -> 8 qwords (16 dwords = 32 halves), amount 3 -> 4 dwords.
  const uint32_t asb = (uint32_t)(uintptr_t)(void*)As;
  const uint32_t bsb = (uint32_t)(uintptr_t)(void*)Bs;
  if (wave == 0) {
    tdm_load_2d(A + rowBase * K, asb, K, M, BK, BM, 3, 3);
    tdm_load_2d(W + colBase * K, bsb, K, N, BK, BN, 3, 3);
  }
  for (int it = 0; it < niter; ++it) {
    const int buf = it & 1;
    if (wave == 0) __builtin_amdgcn_s_wait_tensorcnt(0);
    __syncthreads();
    if (wave == 0 && it + 1 < niter) {
      const int nb = 1 - buf;
      const int k0 = (it + 1) * BK;
      tdm_load_2d(A + rowBase * K + k0, asb + (uint32_t)(nb * BM * AST * 2), K, M, BK, BM, 3, 3);
      tdm_load_2d(W + colBase * K + k0, bsb + (uint32_t)(nb * BN * AST * 2), K, N, BK, BN, 3, 3);
    }
    const _Float16* at = &As[buf * BM * AST];
    const _Float16* bt = &Bs[buf * BN * AST];
    v16h af[4];
#pragma unroll
    for (int mt = 0; mt < 4; ++mt)
      af[mt] = frag_ld_a(at + (wm * 64 + mt * 16) * AST, lane, AST);
#pragma unroll
    for (int nt = 0; nt < 4; ++nt) {
      v16h bf = frag_ld_b(bt + (wn * 64 + nt * 16) * AST, lane, AST);
#pragma unroll
      for (int mt = 0; mt < 4; ++mt) acc[mt][nt] = WMMA_F16(af[mt], bf, acc[mt][nt]);
    }
    // buffer reuse is protected by the wait+barrier at the top of it+1.
  }
#else
  for (int it = 0; it < niter; ++it) {
    const int k0 = it * BK;
    __syncthreads();
#pragma unroll
    for (int i = 0; i < 2; ++i) {  // A: 512 16B segments, 2 per thread
      const int idx = tid + i * 256;
      const int r = idx >> 2, seg = idx & 3;
      *(uint4*)&As[r * AST + seg * 8] =
          *(const uint4*)(A + (rowBase + r) * (size_t)K + k0 + seg * 8);
    }
#pragma unroll
    for (int i = 0; i < 4; ++i) {  // B: 1024 16B segments, 4 per thread
      const int idx = tid + i * 256;
      const int r = idx >> 2, seg = idx & 3;
      *(uint4*)&Bs[r * AST + seg * 8] =
          *(const uint4*)(W + (colBase + r) * (size_t)K + k0 + seg * 8);
    }
    __syncthreads();
    v16h af[4];
#pragma unroll
    for (int mt = 0; mt < 4; ++mt)
      af[mt] = frag_ld_a(&As[(wm * 64 + mt * 16) * AST], lane, AST);
#pragma unroll
    for (int nt = 0; nt < 4; ++nt) {
      v16h bf = frag_ld_b(&Bs[(wn * 64 + nt * 16) * AST], lane, AST);
#pragma unroll
      for (int mt = 0; mt < 4; ++mt) acc[mt][nt] = WMMA_F16(af[mt], bf, acc[mt][nt]);
    }
  }
#endif

  // Epilogue.  C/D layout: lanes 0-15 -> M = vgpr, lanes 16-31 -> M = vgpr+8.
  const int halfSel = (lane >> 4) & 1;
  const int nn = lane & 15;
#pragma unroll
  for (int mt = 0; mt < 4; ++mt) {
#pragma unroll
    for (int nt = 0; nt < 4; ++nt) {
      const size_t gn = colBase + wn * 64 + nt * 16 + nn;
      const float bv = bias[gn];
#pragma unroll
      for (int v = 0; v < 8; ++v) {
        const size_t gm = rowBase + wm * 64 + mt * 16 + v + 8 * halfSel;
        float val = acc[mt][nt][v] + bv;
        if (RELU) val = fmaxf(val, 0.0f);
        if (RES) val += res[gm * (size_t)N + gn];
        if (OUT16) out16[gm * (size_t)N + gn] = (_Float16)val;
        else       out32[gm * (size_t)N + gn] = val;
      }
    }
  }
}

// ---------------------------------------------------------------------------
// Flash attention.  qkv f16 [B,S,3*HS]; block = (b, h, 64-query tile);
// 4 waves x 16 query rows; 64-key blocks; online softmax in registers.
// K tile staged by TDM (pad_interval 32 dwords + 4 dwords pad => KST stride);
// V staged manually (transposed scatter, TDM cannot transpose).
// ---------------------------------------------------------------------------
#define KST 72  // padded LDS row stride (halves): 144B, 16B-aligned

__global__ __launch_bounds__(128) void flash_attn(const _Float16* __restrict__ qkv,
                                                  const float* __restrict__ mask,
                                                  _Float16* __restrict__ ctx, int S) {
  __shared__ __align__(16) _Float16 Ks[64 * KST];      // [key][dim]
  __shared__ __align__(16) _Float16 Vs[64 * KST];      // [dim][key] (transposed)
  __shared__ __align__(16) _Float16 Ps[4][16 * KST];   // per-wave P scratch

  const int bh = blockIdx.x;
  const int b = bh / NHEAD, h = bh % NHEAD;
  const int q0 = blockIdx.y * 64;
  const int tid = threadIdx.x;
  const int lane = tid & 31;
  const int wave = tid >> 5;
  const int halfSel = (lane >> 4) & 1;
  const int nn = lane & 15;
  const size_t tok = 3 * HS;
  const _Float16* qbase = qkv + (size_t)b * S * tok + h * HEAD_DIM;
  const _Float16* kbase = qbase + HS;
  const _Float16* vbase = qbase + 2 * HS;
  const float* mrow = mask + (size_t)b * S;
  const float scale = 0.125f;  // 1/sqrt(64)

  const _Float16* qtile = qbase + (size_t)(q0 + wave * 16) * tok;
  v16h qf0 = frag_ld_a(qtile, lane, (int)tok);
  v16h qf1 = frag_ld_a(qtile + 32, lane, (int)tok);

#if USE_TDM
  const uint32_t ksb = (uint32_t)(uintptr_t)(void*)Ks;
#endif

  float row_m[8], row_l[8];
  v8f oacc[4];
#pragma unroll
  for (int v = 0; v < 8; ++v) { row_m[v] = -1e30f; row_l[v] = 0.0f; }
#pragma unroll
  for (int j = 0; j < 4; ++j)
#pragma unroll
    for (int v = 0; v < 8; ++v) oacc[j][v] = 0.0f;

  for (int kb0 = 0; kb0 < S; kb0 += 64) {
    __syncthreads();
#if USE_TDM
    // K tile: 64 rows x 64 halves; pad_interval code 4 (32 dwords), amount 4 dw.
    if (wave == 0)
      tdm_load_2d(kbase + (size_t)kb0 * tok, ksb, (uint32_t)tok, (uint32_t)S, 64, 64, 4, 3);
#pragma unroll
    for (int i = 0; i < 4; ++i) {
      const int idx = tid + i * 128;
      const int r = idx >> 3, seg = idx & 7;
      const _Float16* src = vbase + (size_t)(kb0 + r) * tok + seg * 8;
#pragma unroll
      for (int d = 0; d < 8; ++d) Vs[(seg * 8 + d) * KST + r] = src[d];
    }
    if (wave == 0) __builtin_amdgcn_s_wait_tensorcnt(0);
#else
#pragma unroll
    for (int i = 0; i < 4; ++i) {
      const int idx = tid + i * 128;
      const int r = idx >> 3, seg = idx & 7;
      *(uint4*)&Ks[r * KST + seg * 8] =
          *(const uint4*)(kbase + (size_t)(kb0 + r) * tok + seg * 8);
      const _Float16* src = vbase + (size_t)(kb0 + r) * tok + seg * 8;
#pragma unroll
      for (int d = 0; d < 8; ++d) Vs[(seg * 8 + d) * KST + r] = src[d];
    }
#endif
    __syncthreads();

    v8f sacc[4];
#pragma unroll
    for (int nt = 0; nt < 4; ++nt) {
#pragma unroll
      for (int v = 0; v < 8; ++v) sacc[nt][v] = 0.0f;
      sacc[nt] = WMMA_F16(qf0, frag_ld_b(&Ks[nt * 16 * KST], lane, KST), sacc[nt]);
      sacc[nt] = WMMA_F16(qf1, frag_ld_b(&Ks[nt * 16 * KST + 32], lane, KST), sacc[nt]);
    }

    float madd[4];
#pragma unroll
    for (int nt = 0; nt < 4; ++nt) madd[nt] = mrow[kb0 + nt * 16 + nn] * -1e8f;

    float p[4][8];
#pragma unroll
    for (int v = 0; v < 8; ++v) {
      float mx = row_m[v];
#pragma unroll
      for (int nt = 0; nt < 4; ++nt) mx = fmaxf(mx, sacc[nt][v] * scale + madd[nt]);
#pragma unroll
      for (int off = 1; off < 16; off <<= 1) mx = fmaxf(mx, __shfl_xor(mx, off, 32));
      const float alpha = __expf(row_m[v] - mx);
      float lsum = 0.0f;
#pragma unroll
      for (int nt = 0; nt < 4; ++nt) {
        const float pv = __expf(sacc[nt][v] * scale + madd[nt] - mx);
        p[nt][v] = pv;
        lsum += pv;
      }
#pragma unroll
      for (int off = 1; off < 16; off <<= 1) lsum += __shfl_xor(lsum, off, 32);
      row_l[v] = row_l[v] * alpha + lsum;
      row_m[v] = mx;
#pragma unroll
      for (int j = 0; j < 4; ++j) oacc[j][v] *= alpha;
    }

#pragma unroll
    for (int nt = 0; nt < 4; ++nt)
#pragma unroll
      for (int v = 0; v < 8; ++v)
        Ps[wave][(v + 8 * halfSel) * KST + nt * 16 + nn] = (_Float16)p[nt][v];
    __syncthreads();

    v16h pf0 = frag_ld_a(&Ps[wave][0], lane, KST);
    v16h pf1 = frag_ld_a(&Ps[wave][32], lane, KST);
#pragma unroll
    for (int j = 0; j < 4; ++j) {
      oacc[j] = WMMA_F16(pf0, frag_ld_b(&Vs[j * 16 * KST], lane, KST), oacc[j]);
      oacc[j] = WMMA_F16(pf1, frag_ld_b(&Vs[j * 16 * KST + 32], lane, KST), oacc[j]);
    }
  }

#pragma unroll
  for (int j = 0; j < 4; ++j)
#pragma unroll
    for (int v = 0; v < 8; ++v) {
      const size_t m = q0 + wave * 16 + v + 8 * halfSel;
      ctx[((size_t)b * S + m) * HS + h * HEAD_DIM + j * 16 + nn] =
          (_Float16)(oacc[j][v] / row_l[v]);
    }
}

// ---------------------------------------------------------------------------
extern "C" void kernel_launch(void* const* d_in, const int* in_sizes, int n_in,
                              void* d_out, int out_size, void* d_ws, size_t ws_size,
                              hipStream_t stream) {
  (void)n_in; (void)out_size; (void)ws_size;
  const float* x        = (const float*)d_in[0];
  const float* pad_mask = (const float*)d_in[1];
  const float* qkv_w    = (const float*)d_in[2];
  const float* qkv_b    = (const float*)d_in[3];
  const float* ao_w     = (const float*)d_in[4];
  const float* ao_b     = (const float*)d_in[5];
  const float* ln1_g    = (const float*)d_in[6];
  const float* ln1_b    = (const float*)d_in[7];
  const float* in_w     = (const float*)d_in[8];
  const float* in_b     = (const float*)d_in[9];
  const float* fo_w     = (const float*)d_in[10];
  const float* fo_b     = (const float*)d_in[11];
  const float* ln2_g    = (const float*)d_in[12];
  const float* ln2_b    = (const float*)d_in[13];
  float* out = (float*)d_out;

  const int M = in_sizes[0] / HS;  // B*S tokens (8192)
  const int S = 1024;
  const int bs = M / S;

  char* ws = (char*)d_ws;
  size_t ofs = 0;
  auto carve = [&](size_t bytes) -> char* {
    ofs = (ofs + 255) & ~(size_t)255;
    char* p = ws + ofs;
    ofs += bytes;
    return p;
  };
  _Float16* wq16  = (_Float16*)carve((size_t)3 * HS * HS * 2);
  _Float16* wo16  = (_Float16*)carve((size_t)HS * HS * 2);
  _Float16* wi16  = (_Float16*)carve((size_t)IMS * HS * 2);
  _Float16* wf16  = (_Float16*)carve((size_t)HS * IMS * 2);
  _Float16* act16 = (_Float16*)carve((size_t)M * HS * 2);   // x_ln, later y_ln
  _Float16* big16 = (_Float16*)carve((size_t)M * IMS * 2);  // qkv, later h
  _Float16* ctx16 = (_Float16*)carve((size_t)M * HS * 2);
  float*    y32   = (float*)carve((size_t)M * HS * 4);

  cvt_f32_to_f16<<<2048, 256, 0, stream>>>(qkv_w, wq16, (size_t)3 * HS * HS);
  cvt_f32_to_f16<<<2048, 256, 0, stream>>>(ao_w, wo16, (size_t)HS * HS);
  cvt_f32_to_f16<<<2048, 256, 0, stream>>>(in_w, wi16, (size_t)IMS * HS);
  cvt_f32_to_f16<<<2048, 256, 0, stream>>>(fo_w, wf16, (size_t)HS * IMS);

  ln_to_f16<<<M, 256, 0, stream>>>(x, ln1_g, ln1_b, act16);

  gemm_f16<false, false, true><<<dim3(3 * HS / BN, M / BM), 256, 0, stream>>>(
      act16, wq16, qkv_b, nullptr, big16, nullptr, M, 3 * HS, HS);

  flash_attn<<<dim3(bs * NHEAD, S / 64), 128, 0, stream>>>(big16, pad_mask, ctx16, S);

  gemm_f16<false, true, false><<<dim3(HS / BN, M / BM), 256, 0, stream>>>(
      ctx16, wo16, ao_b, x, nullptr, y32, M, HS, HS);

  ln_to_f16<<<M, 256, 0, stream>>>(y32, ln2_g, ln2_b, act16);

  gemm_f16<true, false, true><<<dim3(IMS / BN, M / BM), 256, 0, stream>>>(
      act16, wi16, in_b, nullptr, big16, nullptr, M, IMS, HS);

  gemm_f16<false, true, false><<<dim3(HS / BN, M / BM), 256, 0, stream>>>(
      big16, wf16, fo_b, y32, nullptr, out, M, HS, IMS);
}
